// AttentionLayer_50654844289672
// MI455X (gfx1250) — matrix operand order repaired and checked by reading, not compiled, not run on hardware
//
#include <hip/hip_runtime.h>

// Sliding-window attention (window=512, C=32, T=65536) for MI455X / gfx1250.
// Phase 1: fp32 q/k/v projections -> bf16 in workspace (12.6 MB, L2-resident).
// Phase 2: flash-style windowed attention, all GEMMs via v_wmma_f32_16x16x32_bf16.
//          Softmax runs in the exp2 domain (log2e folded into the q scale).
//          Structure guarantees the interior loop carries no mask/select code:
//          band-partial chunks (it=0, it=16) are peeled, sequence-edge waves
//          take a separate fully-masked loop.

#define T_LEN 65536
#define CDIM  32
#define QDIM  64
#define ODIM  64

typedef __attribute__((ext_vector_type(16))) __bf16   v16bf;
typedef __attribute__((ext_vector_type(8)))  float    v8f;
typedef __attribute__((ext_vector_type(4)))  unsigned u32x4;
typedef __attribute__((ext_vector_type(4)))  float    f32x4;

union BF16x16 { v16bf v; u32x4 i4[2]; unsigned u[8]; __bf16 h[16]; };

static __device__ __forceinline__ int iclamp(int x, int lo, int hi) {
  return x < lo ? lo : (x > hi ? hi : x);
}
static __device__ __forceinline__ unsigned pk2(float a, float b) {
  union { __bf16 h[2]; unsigned u; } z;
  z.h[0] = (__bf16)a; z.h[1] = (__bf16)b;
  return z.u;
}

// ---------------------------------------------------------------------------
// Phase 1: q = (log2e/sqrt(32))*(qw@x1+qb), k = kw@x1+kb, v = vw@x2+vb -> bf16.
// qg,kg layout: [t][c] (c contiguous)  -> WMMA A/B fragments are 16B chunks.
// vg   layout: [c][t] (t contiguous)  -> V A-fragments are 16B chunks.
// ---------------------------------------------------------------------------
__global__ void __launch_bounds__(256) qkv_proj_kernel(
    const float* __restrict__ x1, const float* __restrict__ x2,
    const float* __restrict__ qw, const float* __restrict__ qb,
    const float* __restrict__ kw, const float* __restrict__ kbi,
    const float* __restrict__ vw, const float* __restrict__ vb,
    const float* __restrict__ ow,
    __bf16* __restrict__ qg, __bf16* __restrict__ kg,
    __bf16* __restrict__ vg, __bf16* __restrict__ owg)
{
  __shared__ float wq[CDIM * QDIM], wk[CDIM * QDIM], wv[CDIM * QDIM];
  __shared__ float bqs[CDIM], bks[CDIM], bvs[CDIM];
  const int tid = threadIdx.x;
  for (int i = tid; i < CDIM * QDIM; i += 256) { wq[i] = qw[i]; wk[i] = kw[i]; wv[i] = vw[i]; }
  if (tid < CDIM) { bqs[tid] = qb[tid]; bks[tid] = kbi[tid]; bvs[tid] = vb[tid]; }
  if (blockIdx.x == 0)
    for (int i = tid; i < ODIM * CDIM; i += 256) owg[i] = (__bf16)ow[i];
  __syncthreads();

  const int t = blockIdx.x * 256 + tid;
  float xa[QDIM], xb[QDIM];
#pragma unroll
  for (int i = 0; i < QDIM; ++i) {
    xa[i] = x1[(size_t)i * T_LEN + t];
    xb[i] = x2[(size_t)i * T_LEN + t];
  }
  // 1/sqrt(32) * log2(e): softmax runs in the exp2 domain downstream
  const float scale = 0.25504379191899743f;
  union { __bf16 h[CDIM]; u32x4 q[4]; } uq, uk;
#pragma unroll 4
  for (int c = 0; c < CDIM; ++c) {
    float aq = bqs[c], ak = bks[c], av = bvs[c];
#pragma unroll
    for (int i = 0; i < QDIM; ++i) {
      aq = fmaf(wq[c * QDIM + i], xa[i], aq);
      ak = fmaf(wk[c * QDIM + i], xa[i], ak);
      av = fmaf(wv[c * QDIM + i], xb[i], av);
    }
    uq.h[c] = (__bf16)(aq * scale);
    uk.h[c] = (__bf16)ak;
    vg[(size_t)c * T_LEN + t] = (__bf16)av;
  }
  u32x4* qo = (u32x4*)(qg + (size_t)t * CDIM);
  u32x4* ko = (u32x4*)(kg + (size_t)t * CDIM);
#pragma unroll
  for (int j = 0; j < 4; ++j) { qo[j] = uq.q[j]; ko[j] = uk.q[j]; }
}

// ---------------------------------------------------------------------------
// One flash step over a 32-key chunk.
//   MODE 0: interior — no masks, no clamps, prefetch next chunk
//   MODE 1: band-partial (it=0/16 of interior waves) — band mask only
//   MODE 2: sequence-edge waves — band + bounds masks, clamped addresses
// lrow holds only this lane's half of the denominator (pair-combined at end).
// ---------------------------------------------------------------------------
template <int MODE>
static __device__ __forceinline__ void flash_step(
    int kb, int col, bool hi, int off, int qabs,
    const __bf16* __restrict__ kg, const __bf16* __restrict__ vg,
    const BF16x16& Bq, v8f& acc0, v8f& acc1, float& mrow, float& lrow)
{
  const v8f zero = {0, 0, 0, 0, 0, 0, 0, 0};
  int kr0 = kb + col;                 // K A-operand rows (keys)
  int kr1 = kb + 16 + col;
  int ka0 = kb + (hi ? 16 : 0);       // V A-operand key-chunk starts
  int ka1 = kb + 8 + (hi ? 16 : 0);
  if (MODE == 2) {                    // clamped reads stay in-bounds; finite
    kr0 = iclamp(kr0, 0, T_LEN - 1);  // garbage is masked to -1e30 / * p==0
    kr1 = iclamp(kr1, 0, T_LEN - 1);
    ka0 = iclamp(ka0, 0, T_LEN - 8);
    ka1 = iclamp(ka1, 0, T_LEN - 8);
  }
  BF16x16 Ak0, Ak1;
  Ak0.i4[0] = *(const u32x4*)(kg + (size_t)kr0 * CDIM + (hi ? 8 : 0));
  Ak0.i4[1] = *(const u32x4*)(kg + (size_t)kr0 * CDIM + 16 + (hi ? 8 : 0));
  Ak1.i4[0] = *(const u32x4*)(kg + (size_t)kr1 * CDIM + (hi ? 8 : 0));
  Ak1.i4[1] = *(const u32x4*)(kg + (size_t)kr1 * CDIM + 16 + (hi ? 8 : 0));
  if (MODE == 0)  // next chunk is provably in-range for interior waves
    __builtin_prefetch(kg + (size_t)(kb + 32 + col) * CDIM, 0, 1);

  v8f e0 = __builtin_amdgcn_wmma_f32_16x16x32_bf16(false, Ak0.v, false, Bq.v, (short)0, zero, false, false);
  v8f e1 = __builtin_amdgcn_wmma_f32_16x16x32_bf16(false, Ak1.v, false, Bq.v, (short)0, zero, false, false);

  float ev[16];
  if (MODE == 0) {
#pragma unroll
    for (int i = 0; i < 8; ++i) { ev[i] = e0[i]; ev[i + 8] = e1[i]; }
  } else {
    // band (and for MODE 2 bounds) via unsigned-range trick;
    // -1e30 junk self-cancels later through r = exp2(-1e30 - m_real) = 0
    const int dbase = kb + off - (qabs - 256);
#pragma unroll
    for (int i = 0; i < 8; ++i) {
      bool v0 = ((unsigned)(dbase + i)      < 512u);
      bool v1 = ((unsigned)(dbase + i + 16) < 512u);
      if (MODE == 2) {
        const int k0 = kb + i + off;
        v0 = v0 & ((unsigned)k0        < (unsigned)T_LEN);
        v1 = v1 & ((unsigned)(k0 + 16) < (unsigned)T_LEN);
      }
      ev[i]     = v0 ? e0[i] : -1e30f;
      ev[i + 8] = v1 ? e1[i] : -1e30f;
    }
  }

  float tmax = ev[0];
#pragma unroll
  for (int i = 1; i < 16; ++i) tmax = fmaxf(tmax, ev[i]);
  tmax = fmaxf(tmax, __shfl_xor(tmax, 16, 32));  // pair must agree on m
  const float mnew = fmaxf(mrow, tmax);
  const float r = __builtin_amdgcn_exp2f(mrow - mnew);  // logits are log2-scaled
  mrow = mnew;
  float p[16], s = 0.f;
#pragma unroll
  for (int i = 0; i < 16; ++i) { p[i] = __builtin_amdgcn_exp2f(ev[i] - mnew); s += p[i]; }
  lrow = lrow * r + s;                // local half only; combined after loop
#pragma unroll
  for (int i = 0; i < 8; ++i) { acc0[i] *= r; acc1[i] *= r; }

  // repack P (E^T D-layout) into B-operand layout: one half-swap shuffle
  unsigned t0p[4], t1p[4];
#pragma unroll
  for (int j = 0; j < 4; ++j) {
    t0p[j] = pk2(p[2 * j], p[2 * j + 1]);
    t1p[j] = pk2(p[8 + 2 * j], p[9 + 2 * j]);
  }
  BF16x16 Bp;
#pragma unroll
  for (int j = 0; j < 4; ++j) {
    const unsigned sendv = hi ? t0p[j] : t1p[j];
    const unsigned recv  = (unsigned)__shfl_xor((int)sendv, 16, 32);
    Bp.u[j]     = hi ? recv   : t0p[j];
    Bp.u[j + 4] = hi ? t1p[j] : recv;
  }

  BF16x16 Av0, Av1;
  Av0.i4[0] = *(const u32x4*)(vg + (size_t)col * T_LEN + ka0);
  Av0.i4[1] = *(const u32x4*)(vg + (size_t)col * T_LEN + ka1);
  Av1.i4[0] = *(const u32x4*)(vg + (size_t)(col + 16) * T_LEN + ka0);
  Av1.i4[1] = *(const u32x4*)(vg + (size_t)(col + 16) * T_LEN + ka1);
  acc0 = __builtin_amdgcn_wmma_f32_16x16x32_bf16(false, Av0.v, false, Bp.v, (short)0, acc0, false, false);
  acc1 = __builtin_amdgcn_wmma_f32_16x16x32_bf16(false, Av1.v, false, Bp.v, (short)0, acc1, false, false);
}

// ---------------------------------------------------------------------------
// Phase 2: per wave: 16 queries. 17 chunks of 32 keys cover [q-256, q+256).
//  E^T (keys x queries) = K_tile(A) x Q(B)   -> softmax stays in lane pairs
//  out^T (c x queries) += V_tile(A) x P^T(B)
//  y (queries x 64)     = relu(out)(A, == out^T D regs!) x ow^T(B), +ob, *mask
// ---------------------------------------------------------------------------
__global__ void __launch_bounds__(256) flash_attn_kernel(
    const __bf16* __restrict__ qg, const __bf16* __restrict__ kg,
    const __bf16* __restrict__ vg, const __bf16* __restrict__ owg,
    const float* __restrict__ ob, const float* __restrict__ mask,
    float* __restrict__ out)
{
  const int  lane = threadIdx.x & 31;
  const int  wid  = threadIdx.x >> 5;
  const int  col  = lane & 15;
  const bool hi   = lane >= 16;
  const int  off  = hi ? 8 : 0;
  const int  qs   = blockIdx.x * 128 + wid * 16;
  const int  qabs = qs + col;

  // Q as B-operand (K=c=32, N=16 queries): lane<16 -> c 0..7,16..23, lane>=16 -> c 8..15,24..31
  BF16x16 Bq;
  {
    const u32x4* p = (const u32x4*)(qg + (size_t)(qs + col) * CDIM + (hi ? 8 : 0));
    Bq.i4[0] = p[0]; Bq.i4[1] = p[2];
  }

  v8f acc0 = {0, 0, 0, 0, 0, 0, 0, 0};
  v8f acc1 = {0, 0, 0, 0, 0, 0, 0, 0};
  float mrow = -1e30f, lrow = 0.f;
  const int kb0 = (qs - 256) & ~31;

  // wave-level split: interior waves never touch sequence bounds
  if ((kb0 >= 0) && (kb0 + 544 <= T_LEN)) {
    flash_step<1>(kb0, col, hi, off, qabs, kg, vg, Bq, acc0, acc1, mrow, lrow);
    for (int it = 1; it < 16; ++it)   // hot loop: no compares, no selects
      flash_step<0>(kb0 + it * 32, col, hi, off, qabs, kg, vg, Bq, acc0, acc1, mrow, lrow);
    flash_step<1>(kb0 + 512, col, hi, off, qabs, kg, vg, Bq, acc0, acc1, mrow, lrow);
  } else {
    for (int it = 0; it < 17; ++it)
      flash_step<2>(kb0 + it * 32, col, hi, off, qabs, kg, vg, Bq, acc0, acc1, mrow, lrow);
  }
  lrow += __shfl_xor(lrow, 16, 32);   // combine the pair's denominator halves

  // normalize + relu; D-layout of out^T IS the A-layout of out for the o-proj
  const float inv = 1.0f / lrow;
  BF16x16 Ao;
#pragma unroll
  for (int j = 0; j < 4; ++j) {
    Ao.u[j]     = pk2(fmaxf(acc0[2 * j] * inv, 0.f), fmaxf(acc0[2 * j + 1] * inv, 0.f));
    Ao.u[j + 4] = pk2(fmaxf(acc1[2 * j] * inv, 0.f), fmaxf(acc1[2 * j + 1] * inv, 0.f));
  }

  const v8f zero = {0, 0, 0, 0, 0, 0, 0, 0};
  const int tbase = qs + off;
  const f32x4 m0 = *(const f32x4*)(mask + tbase);      // pmask = mask row 0
  const f32x4 m1 = *(const f32x4*)(mask + tbase + 4);
#pragma unroll
  for (int ot = 0; ot < 4; ++ot) {
    BF16x16 Bo;  // ow^T as B-operand: owg is [o][c], c contiguous
    const u32x4* p = (const u32x4*)(owg + (size_t)(ot * 16 + col) * CDIM + (hi ? 8 : 0));
    Bo.i4[0] = p[0]; Bo.i4[1] = p[2];
    v8f y = __builtin_amdgcn_wmma_f32_16x16x32_bf16(false, Ao.v, false, Bo.v, (short)0, zero, false, false);
    const int o = ot * 16 + col;
    const float bias = ob[o];
    f32x4 r0, r1;
#pragma unroll
    for (int i = 0; i < 4; ++i) {
      r0[i] = (y[i] + bias) * m0[i];
      r1[i] = (y[i + 4] + bias) * m1[i];
    }
    *(f32x4*)(out + (size_t)o * T_LEN + tbase)     = r0;
    *(f32x4*)(out + (size_t)o * T_LEN + tbase + 4) = r1;
  }
}

// ---------------------------------------------------------------------------
extern "C" void kernel_launch(void* const* d_in, const int* in_sizes, int n_in,
                              void* d_out, int out_size, void* d_ws, size_t ws_size,
                              hipStream_t stream) {
  const float* x1   = (const float*)d_in[0];
  const float* x2   = (const float*)d_in[1];
  const float* mask = (const float*)d_in[2];
  // d_in[3] = dilation (512, hardcoded)
  const float* qw = (const float*)d_in[4];
  const float* qb = (const float*)d_in[5];
  const float* kw = (const float*)d_in[6];
  const float* kb = (const float*)d_in[7];
  const float* vw = (const float*)d_in[8];
  const float* vb = (const float*)d_in[9];
  const float* ow = (const float*)d_in[10];
  const float* ob = (const float*)d_in[11];

  const size_t QKV_BYTES = (size_t)T_LEN * CDIM * sizeof(__bf16);  // 4 MB each
  char* ws = (char*)d_ws;
  __bf16* qg  = (__bf16*)(ws);
  __bf16* kg  = (__bf16*)(ws + QKV_BYTES);
  __bf16* vg  = (__bf16*)(ws + 2 * QKV_BYTES);
  __bf16* owg = (__bf16*)(ws + 3 * QKV_BYTES);  // 4 KB; total ws ~12.6 MB

  qkv_proj_kernel<<<T_LEN / 256, 256, 0, stream>>>(x1, x2, qw, qb, kw, kb, vw, vb,
                                                   ow, qg, kg, vg, owg);
  flash_attn_kernel<<<T_LEN / 128, 256, 0, stream>>>(qg, kg, vg, owg, ob, mask,
                                                     (float*)d_out);
}
